// StreamingConformerBlock_63814624084455
// MI455X (gfx1250) — matrix operand order repaired and checked by reading, not compiled, not run on hardware
//
#include <hip/hip_runtime.h>
#include <hip/hip_bf16.h>
#include <math.h>

typedef __attribute__((ext_vector_type(16))) __bf16 bf16x16;
typedef __attribute__((ext_vector_type(8)))  float  f32x8;
typedef unsigned short u16;
typedef unsigned int   u32;

union Frag { bf16x16 v; uint4 q[2]; };

__device__ __forceinline__ u16 f2bf(float f) {
  u32 x = __float_as_uint(f);
  u32 r = x + 0x7FFFu + ((x >> 16) & 1u);   // round-to-nearest-even
  return (u16)(r >> 16);
}

// ---- DPP16 cross-lane reductions (stay within each 16-lane row, pure VALU) --
template <int CTRL>
__device__ __forceinline__ float dppf(float v) {
  return __int_as_float(
      __builtin_amdgcn_update_dpp(0, __float_as_int(v), CTRL, 0xF, 0xF, true));
}
__device__ __forceinline__ float rowmax16(float v) {
  v = fmaxf(v, dppf<0xB1>(v));    // quad_perm(1,0,3,2): xor1
  v = fmaxf(v, dppf<0x4E>(v));    // quad_perm(2,3,0,1): xor2
  v = fmaxf(v, dppf<0x141>(v));   // row_half_mirror : xor7
  v = fmaxf(v, dppf<0x140>(v));   // row_mirror      : xor15
  return v;
}
__device__ __forceinline__ float rowsum16(float v) {
  v += dppf<0xB1>(v);
  v += dppf<0x4E>(v);
  v += dppf<0x141>(v);
  v += dppf<0x140>(v);
  return v;
}

// ---------------------------------------------------------------------------
// Weight convert + transpose: W[K,N] f32 -> Wt[N,K] bf16
// ---------------------------------------------------------------------------
__global__ __launch_bounds__(256) void convert_w_t(const float* __restrict__ W,
                                                   u16* __restrict__ Wt, int K, int N) {
  int i = blockIdx.x * 256 + threadIdx.x;
  if (i >= K * N) return;
  int k = i / N, n = i % N;
  Wt[(size_t)n * K + k] = f2bf(W[i]);
}

__global__ __launch_bounds__(256) void concat3(const float* __restrict__ a,
                                               const float* __restrict__ b,
                                               const float* __restrict__ c,
                                               float* __restrict__ o) {
  int i = blockIdx.x * 256 + threadIdx.x;   // 1536
  o[i] = (i < 512) ? a[i] : ((i < 1024) ? b[i - 512] : c[i - 1024]);
}

// ---------------------------------------------------------------------------
// LayerNorm (eps=1e-3): one wave per 512-wide row, bf16 output
// ---------------------------------------------------------------------------
__global__ __launch_bounds__(256) void ln_bf16(const float* __restrict__ x,
                                               const float* __restrict__ g,
                                               const float* __restrict__ b,
                                               u16* __restrict__ out) {
  const int row  = blockIdx.x * 8 + (threadIdx.x >> 5);
  const int lane = threadIdx.x & 31;
  const float* xr = x + (size_t)row * 512 + lane * 16;
  float4 v[4];
#pragma unroll
  for (int i = 0; i < 4; ++i) v[i] = ((const float4*)xr)[i];
  float s = 0.f, s2 = 0.f;
#pragma unroll
  for (int i = 0; i < 4; ++i) {
    s  += v[i].x + v[i].y + v[i].z + v[i].w;
    s2 += v[i].x * v[i].x + v[i].y * v[i].y + v[i].z * v[i].z + v[i].w * v[i].w;
  }
  s  = rowsum16(s);  s  += __shfl_xor(s, 16, 32);
  s2 = rowsum16(s2); s2 += __shfl_xor(s2, 16, 32);
  const float mean = s * (1.0f / 512.0f);
  const float var  = s2 * (1.0f / 512.0f) - mean * mean;
  const float rst  = rsqrtf(var + 1e-3f);
  union { u16 h[16]; uint4 q[2]; } ou;
  const float* gp = g + lane * 16;
  const float* bp = b + lane * 16;
#pragma unroll
  for (int i = 0; i < 4; ++i) {
    float4 gg = ((const float4*)gp)[i];
    float4 bb = ((const float4*)bp)[i];
    ou.h[i * 4 + 0] = f2bf((v[i].x - mean) * rst * gg.x + bb.x);
    ou.h[i * 4 + 1] = f2bf((v[i].y - mean) * rst * gg.y + bb.y);
    ou.h[i * 4 + 2] = f2bf((v[i].z - mean) * rst * gg.z + bb.z);
    ou.h[i * 4 + 3] = f2bf((v[i].w - mean) * rst * gg.w + bb.w);
  }
  uint4* op = (uint4*)(out + (size_t)row * 512 + lane * 16);
  op[0] = ou.q[0];
  op[1] = ou.q[1];
}

// ---------------------------------------------------------------------------
// WMMA GEMM: C[M,N] = A[M,K](bf16) * Bt[N,K](bf16)^T + bias, fused epilogue.
// Block tile 128x128, BK=64, 8 waves (2x4), 64x32 per wave.
// Double-buffered LDS (72 KB), rows padded to 72 elems -> conflict-free b128.
// ---------------------------------------------------------------------------
enum { EPI_F32 = 0, EPI_RELU_BF16 = 1, EPI_BIAS_BF16 = 2, EPI_GELU_F32 = 3,
       EPI_RES_F32 = 4, EPI_COMBINE_F32 = 5 };

template <int EPI>
__global__ __launch_bounds__(256) void gemm_wmma(const u16* __restrict__ A, int lda,
                                                 const u16* __restrict__ Bt,
                                                 const float* __restrict__ bias,
                                                 void* __restrict__ Out, int ldc,
                                                 const float* __restrict__ Res,
                                                 const float* __restrict__ Res2,
                                                 int Kd) {
  __shared__ __align__(16) u16 sA[2][128 * 72];
  __shared__ __align__(16) u16 sB[2][128 * 72];
  const int tid    = threadIdx.x;
  const int wave   = tid >> 5, lane = tid & 31;
  const int lane16 = lane & 15, half = lane >> 4;
  const int wm     = (wave >> 2) << 6;   // 0 / 64
  const int wn     = (wave & 3) << 5;    // 0..96
  const size_t bm  = (size_t)blockIdx.y * 128;
  const size_t bn  = (size_t)blockIdx.x * 128;

  f32x8 acc[4][2];
#pragma unroll
  for (int i = 0; i < 4; ++i)
#pragma unroll
    for (int j = 0; j < 2; ++j)
#pragma unroll
      for (int e = 0; e < 8; ++e) acc[i][j][e] = 0.0f;

  const int r = tid >> 1;
  const int c = (tid & 1) << 5;   // 0 or 32 elems within 64-wide k-slab
  const u16* gA = A  + (bm + r) * (size_t)lda + c;
  const u16* gB = Bt + (bn + r) * (size_t)Kd  + c;

  uint4 ra[4], rb[4];
  {  // preload slab 0
    const uint4* ga = (const uint4*)gA;
    const uint4* gb = (const uint4*)gB;
    ra[0] = ga[0]; ra[1] = ga[1]; ra[2] = ga[2]; ra[3] = ga[3];
    rb[0] = gb[0]; rb[1] = gb[1]; rb[2] = gb[2]; rb[3] = gb[3];
    u16* wa = &sA[0][r * 72 + c];
    u16* wb = &sB[0][r * 72 + c];
    ((uint4*)wa)[0] = ra[0]; ((uint4*)wa)[1] = ra[1]; ((uint4*)wa)[2] = ra[2]; ((uint4*)wa)[3] = ra[3];
    ((uint4*)wb)[0] = rb[0]; ((uint4*)wb)[1] = rb[1]; ((uint4*)wb)[2] = rb[2]; ((uint4*)wb)[3] = rb[3];
  }

  int buf = 0;
  for (int k0 = 0; k0 < Kd; k0 += 64) {
    __syncthreads();                       // LDS[buf] ready for all waves
    const bool hasNext = (k0 + 64) < Kd;
    if (hasNext) {                         // issue next slab; consumed after math
      const uint4* ga = (const uint4*)(gA + k0 + 64);
      const uint4* gb = (const uint4*)(gB + k0 + 64);
      ra[0] = ga[0]; ra[1] = ga[1]; ra[2] = ga[2]; ra[3] = ga[3];
      rb[0] = gb[0]; rb[1] = gb[1]; rb[2] = gb[2]; rb[3] = gb[3];
      if (k0 + 128 < Kd) {                 // global_prefetch_b8 two slabs ahead
        __builtin_prefetch(gA + k0 + 128, 0, 1);
        __builtin_prefetch(gB + k0 + 128, 0, 1);
      }
    }

#pragma unroll
    for (int kc = 0; kc < 2; ++kc) {
      Frag fa[4], fb[2];
#pragma unroll
      for (int mt = 0; mt < 4; ++mt) {    // A layout: k chunks at half*8, 16+half*8
        const u16* p = &sA[buf][(wm + mt * 16 + lane16) * 72 + kc * 32];
        fa[mt].q[0] = *(const uint4*)(p + half * 8);
        fa[mt].q[1] = *(const uint4*)(p + 16 + half * 8);
      }
#pragma unroll
      for (int nt = 0; nt < 2; ++nt) {    // B layout: k chunks at half*8, 16+half*8
        const u16* p = &sB[buf][(wn + nt * 16 + lane16) * 72 + kc * 32];
        fb[nt].q[0] = *(const uint4*)(p + half * 8);
        fb[nt].q[1] = *(const uint4*)(p + 16 + half * 8);
      }
#pragma unroll
      for (int mt = 0; mt < 4; ++mt)
#pragma unroll
        for (int nt = 0; nt < 2; ++nt)
          acc[mt][nt] = __builtin_amdgcn_wmma_f32_16x16x32_bf16(
              false, fa[mt].v, false, fb[nt].v, (short)0, acc[mt][nt], false, false);
    }

    if (hasNext) {                         // commit next slab to the other buffer
      u16* wa = &sA[buf ^ 1][r * 72 + c];
      u16* wb = &sB[buf ^ 1][r * 72 + c];
      ((uint4*)wa)[0] = ra[0]; ((uint4*)wa)[1] = ra[1]; ((uint4*)wa)[2] = ra[2]; ((uint4*)wa)[3] = ra[3];
      ((uint4*)wb)[0] = rb[0]; ((uint4*)wb)[1] = rb[1]; ((uint4*)wb)[2] = rb[2]; ((uint4*)wb)[3] = rb[3];
    }
    buf ^= 1;
  }

#pragma unroll
  for (int mt = 0; mt < 4; ++mt) {
#pragma unroll
    for (int nt = 0; nt < 2; ++nt) {
      size_t col  = bn + wn + nt * 16 + lane16;
      float  bcol = bias[col];
#pragma unroll
      for (int e = 0; e < 8; ++e) {       // C layout: row = e + 8*half, col = lane16
        size_t row = bm + wm + mt * 16 + e + 8 * half;
        float  v   = acc[mt][nt][e] + bcol;
        size_t o   = row * (size_t)ldc + col;
        if (EPI == EPI_F32)            ((float*)Out)[o] = v;
        else if (EPI == EPI_RELU_BF16) ((u16*)Out)[o]   = f2bf(fmaxf(v, 0.0f));
        else if (EPI == EPI_BIAS_BF16) ((u16*)Out)[o]   = f2bf(v);
        else if (EPI == EPI_GELU_F32)  ((float*)Out)[o] = 0.5f * v * (1.0f + erff(v * 0.70710678118f));
        else if (EPI == EPI_RES_F32)   ((float*)Out)[o] = Res[o] + v;
        else if (EPI == EPI_COMBINE_F32) ((float*)Out)[o] = Res[o] + 0.5f * Res2[o] + v;
      }
    }
  }
}

// ---------------------------------------------------------------------------
// Tiled V transpose: qkv[B,T,1536] (V section) bf16 -> vt[B,H,HD,T] bf16
// ---------------------------------------------------------------------------
__global__ __launch_bounds__(256) void transpose_v_tiled(const u16* __restrict__ qkv,
                                                         u16* __restrict__ vt) {
  __shared__ __align__(16) u16 tile[64][72];
  const int bh = blockIdx.x >> 4;          // b*8 + h
  const int t0 = (blockIdx.x & 15) << 6;
  const int b  = bh >> 3, h = bh & 7;
  {
    const int tl = threadIdx.x >> 2;
    const int dc = (threadIdx.x & 3) << 4;
    const u16* src = qkv + ((size_t)(b * 1024 + t0 + tl) * 1536) + 1024 + h * 64 + dc;
    *(uint4*)&tile[tl][dc]     = ((const uint4*)src)[0];
    *(uint4*)&tile[tl][dc + 8] = ((const uint4*)src)[1];
  }
  __syncthreads();
  {
    const int d  = threadIdx.x >> 2;
    const int tc = (threadIdx.x & 3) << 4;
    union { u16 h2[16]; uint4 q[2]; } ou;
#pragma unroll
    for (int i = 0; i < 16; ++i) ou.h2[i] = tile[tc + i][d];
    u16* dst = vt + ((size_t)(bh * 64 + d) * 1024) + t0 + tc;
    ((uint4*)dst)[0] = ou.q[0];
    ((uint4*)dst)[1] = ou.q[1];
  }
}

// ---------------------------------------------------------------------------
// Flash attention, causal. One wave per 16-row Q tile, 32 KV rows / iter.
// qkv: [B,T,1536] bf16 (Q|K|V). Vt: [B,H,HD,T] bf16. Ctx: [B,T,512] bf16.
// ---------------------------------------------------------------------------
__global__ __launch_bounds__(256) void flash_attn(const u16* __restrict__ QKV,
                                                  const u16* __restrict__ Vt,
                                                  u16* __restrict__ Ctx) {
  __shared__ __align__(16) u16 pbuf[8][16 * 40];  // padded rows: conflict-free
  const int wave   = threadIdx.x >> 5, lane = threadIdx.x & 31;
  const int lane16 = lane & 15, half = lane >> 4;
  const int qt     = blockIdx.x * 8 + wave;
  const int bh     = blockIdx.y;          // b*H + h
  const int b      = bh >> 3, h = bh & 7;

  Frag fq[2];
  const u16* qrow = QKV + ((size_t)(b * 1024 + qt * 16 + lane16) * 1536) + h * 64;
#pragma unroll
  for (int cc = 0; cc < 2; ++cc) {
    fq[cc].q[0] = *(const uint4*)(qrow + cc * 32 + half * 8);
    fq[cc].q[1] = *(const uint4*)(qrow + cc * 32 + 16 + half * 8);
  }

  f32x8 o[4];
#pragma unroll
  for (int nt = 0; nt < 4; ++nt)
#pragma unroll
    for (int e = 0; e < 8; ++e) o[nt][e] = 0.0f;
  float mrow[8], lrow[8];
#pragma unroll
  for (int e = 0; e < 8; ++e) { mrow[e] = -1e30f; lrow[e] = 0.0f; }

  auto kv_block = [&](int jb, bool masked) {
    Frag fk0[2], fk1[2];
    const u16* kr0 = QKV + ((size_t)(b * 1024 + jb + lane16) * 1536) + 512 + h * 64;
    const u16* kr1 = kr0 + (size_t)16 * 1536;
#pragma unroll
    for (int cc = 0; cc < 2; ++cc) {
      fk0[cc].q[0] = *(const uint4*)(kr0 + cc * 32 + half * 8);
      fk0[cc].q[1] = *(const uint4*)(kr0 + cc * 32 + 16 + half * 8);
      fk1[cc].q[0] = *(const uint4*)(kr1 + cc * 32 + half * 8);
      fk1[cc].q[1] = *(const uint4*)(kr1 + cc * 32 + 16 + half * 8);
    }
    f32x8 s0, s1;
#pragma unroll
    for (int e = 0; e < 8; ++e) { s0[e] = 0.0f; s1[e] = 0.0f; }
    s0 = __builtin_amdgcn_wmma_f32_16x16x32_bf16(false, fq[0].v, false, fk0[0].v, (short)0, s0, false, false);
    s0 = __builtin_amdgcn_wmma_f32_16x16x32_bf16(false, fq[1].v, false, fk0[1].v, (short)0, s0, false, false);
    s1 = __builtin_amdgcn_wmma_f32_16x16x32_bf16(false, fq[0].v, false, fk1[0].v, (short)0, s1, false, false);
    s1 = __builtin_amdgcn_wmma_f32_16x16x32_bf16(false, fq[1].v, false, fk1[1].v, (short)0, s1, false, false);

    float alpha[8];
#pragma unroll
    for (int e = 0; e < 8; ++e) {
      float a0 = s0[e] * 0.125f;          // 1/sqrt(64)
      float a1 = s1[e] * 0.125f;
      if (masked) {
        int rowi = (qt << 4) + e + (half << 3);
        if (jb + lane16 > rowi)      a0 = -1e9f;
        if (jb + 16 + lane16 > rowi) a1 = -1e9f;
      }
      float rm   = rowmax16(fmaxf(a0, a1));
      float mnew = fmaxf(mrow[e], rm);
      float al   = __expf(mrow[e] - mnew);
      float p0   = __expf(a0 - mnew);
      float p1   = __expf(a1 - mnew);
      float rs   = rowsum16(p0 + p1);
      lrow[e]  = lrow[e] * al + rs;
      mrow[e]  = mnew;
      alpha[e] = al;
      pbuf[wave][(e + 8 * half) * 40 + lane16]      = f2bf(p0);
      pbuf[wave][(e + 8 * half) * 40 + 16 + lane16] = f2bf(p1);
    }
#pragma unroll
    for (int nt = 0; nt < 4; ++nt)
#pragma unroll
      for (int e = 0; e < 8; ++e) o[nt][e] *= alpha[e];

    Frag fp;  // P 16x32 as A-layout
    const u16* pp = &pbuf[wave][lane16 * 40];
    fp.q[0] = *(const uint4*)(pp + half * 8);
    fp.q[1] = *(const uint4*)(pp + 16 + half * 8);
#pragma unroll
    for (int nt = 0; nt < 4; ++nt) {
      Frag fv;  // B layout: t chunks at half*8, 16+half*8 at col lane16
      const u16* vp = Vt + ((size_t)(bh * 64 + nt * 16 + lane16) * 1024) + jb;
      fv.q[0] = *(const uint4*)(vp + half * 8);
      fv.q[1] = *(const uint4*)(vp + 16 + half * 8);
      o[nt] = __builtin_amdgcn_wmma_f32_16x16x32_bf16(false, fp.v, false, fv.v, (short)0, o[nt], false, false);
    }
  };

  const int nf = qt >> 1;                  // fully-unmasked 32-blocks
  for (int j = 0; j < nf; ++j) kv_block(j << 5, false);
  kv_block(nf << 5, true);                 // single masked block

#pragma unroll
  for (int e = 0; e < 8; ++e) {
    const float inv = 1.0f / lrow[e];
    const int t = (qt << 4) + e + (half << 3);
    const size_t rowb = ((size_t)(b * 1024 + t) * 512) + h * 64 + lane16;
#pragma unroll
    for (int nt = 0; nt < 4; ++nt) Ctx[rowb + nt * 16] = f2bf(o[nt][e] * inv);
  }
}

// ---------------------------------------------------------------------------
// causal depthwise conv(k=15) + BN(inference) + swish -> bf16 (4 ch / thread)
// ---------------------------------------------------------------------------
__global__ __launch_bounds__(256) void dwconv_bn_swish(const float* __restrict__ in,
    const float* __restrict__ wk, const float* __restrict__ wb,
    const float* __restrict__ bng, const float* __restrict__ bnb,
    const float* __restrict__ bnm, const float* __restrict__ bnv,
    u16* __restrict__ out) {
  size_t idx4 = (size_t)blockIdx.x * 256 + threadIdx.x;  // B*T*256
  const int c4 = (int)(idx4 & 255) << 2;
  const int t  = (int)((idx4 >> 8) & 1023);
  const size_t base = idx4 << 2;
  float ax = 0.f, ay = 0.f, az = 0.f, aw = 0.f;
#pragma unroll
  for (int k = 0; k < 15; ++k) {
    int tt = t + k - 14;
    if (tt >= 0) {
      float4 xv = *(const float4*)(in + base + (long long)(k - 14) * 1024);
      float4 wv = *(const float4*)(wk + k * 1024 + c4);
      ax += xv.x * wv.x; ay += xv.y * wv.y; az += xv.z * wv.z; aw += xv.w * wv.w;
    }
  }
  float4 vb = *(const float4*)(wb + c4);
  float4 vm = *(const float4*)(bnm + c4);
  float4 vv = *(const float4*)(bnv + c4);
  float4 vg = *(const float4*)(bng + c4);
  float4 vbb = *(const float4*)(bnb + c4);
  ax = (ax + vb.x - vm.x) * rsqrtf(vv.x + 1e-3f) * vg.x + vbb.x;
  ay = (ay + vb.y - vm.y) * rsqrtf(vv.y + 1e-3f) * vg.y + vbb.y;
  az = (az + vb.z - vm.z) * rsqrtf(vv.z + 1e-3f) * vg.z + vbb.z;
  aw = (aw + vb.w - vm.w) * rsqrtf(vv.w + 1e-3f) * vg.w + vbb.w;
  ax = ax / (1.0f + __expf(-ax));
  ay = ay / (1.0f + __expf(-ay));
  az = az / (1.0f + __expf(-az));
  aw = aw / (1.0f + __expf(-aw));
  union { u16 h[4]; uint2 q; } r;
  r.h[0] = f2bf(ax); r.h[1] = f2bf(ay); r.h[2] = f2bf(az); r.h[3] = f2bf(aw);
  *(uint2*)(out + base) = r.q;
}

// ---------------------------------------------------------------------------
extern "C" void kernel_launch(void* const* d_in, const int* in_sizes, int n_in,
                              void* d_out, int out_size, void* d_ws, size_t ws_size,
                              hipStream_t stream) {
  (void)in_sizes; (void)n_in; (void)out_size; (void)ws_size;
  const float* x    = (const float*)d_in[0];
  const float* g1   = (const float*)d_in[1];
  const float* b1   = (const float*)d_in[2];
  const float* Wff1 = (const float*)d_in[3];
  const float* bff1 = (const float*)d_in[4];
  const float* Wff2 = (const float*)d_in[5];
  const float* bff2 = (const float*)d_in[6];
  const float* ga   = (const float*)d_in[7];
  const float* ba   = (const float*)d_in[8];
  const float* Wq   = (const float*)d_in[9];
  const float* bq   = (const float*)d_in[10];
  const float* Wk   = (const float*)d_in[11];
  const float* bk   = (const float*)d_in[12];
  const float* Wv   = (const float*)d_in[13];
  const float* bv   = (const float*)d_in[14];
  const float* Wo   = (const float*)d_in[15];
  const float* bo   = (const float*)d_in[16];
  const float* gc   = (const float*)d_in[17];
  const float* bc   = (const float*)d_in[18];
  const float* Wcp  = (const float*)d_in[19];
  const float* bcp  = (const float*)d_in[20];
  const float* dwk  = (const float*)d_in[21];
  const float* dwb  = (const float*)d_in[22];
  const float* bng  = (const float*)d_in[23];
  const float* bnb  = (const float*)d_in[24];
  const float* bnm  = (const float*)d_in[25];
  const float* bnv  = (const float*)d_in[26];
  const float* Wco  = (const float*)d_in[27];
  const float* bco  = (const float*)d_in[28];
  const float* g2   = (const float*)d_in[29];
  const float* b2   = (const float*)d_in[30];
  const float* Wf1  = (const float*)d_in[31];
  const float* bf1  = (const float*)d_in[32];
  const float* Wf2  = (const float*)d_in[33];
  const float* bf2  = (const float*)d_in[34];

  const int NT = 8192;  // B*T tokens
  char* p = (char*)d_ws;
  auto carve = [&](size_t bytes) -> void* {
    char* r = p;
    p += (bytes + 255) & ~(size_t)255;
    return (void*)r;
  };

  u16* wt_ff1 = (u16*)carve((size_t)512 * 2048 * 2);
  u16* wt_ff2 = (u16*)carve((size_t)2048 * 512 * 2);
  u16* wt_qkv = (u16*)carve((size_t)1536 * 512 * 2);   // Q|K|V rows, transposed
  u16* wt_o   = (u16*)carve((size_t)512 * 512 * 2);
  u16* wt_cp  = (u16*)carve((size_t)512 * 1024 * 2);
  u16* wt_co  = (u16*)carve((size_t)1024 * 512 * 2);
  u16* wt_f1  = (u16*)carve((size_t)512 * 2048 * 2);
  u16* wt_f2  = (u16*)carve((size_t)2048 * 512 * 2);
  float* bqkv  = (float*)carve((size_t)1536 * 4);
  u16*   lnbuf = (u16*)carve((size_t)NT * 512 * 2);
  u16*   hidb  = (u16*)carve((size_t)NT * 2048 * 2);   // ffpre / conv / ffpost hidden
  float* f32a  = (float*)carve((size_t)NT * 1024 * 4); // xff (first half), then conv gelu
  u16*   qkvb  = (u16*)carve((size_t)NT * 1536 * 2);
  u16*   vtb   = (u16*)carve((size_t)NT * 512 * 2);
  u16*   ctxb  = (u16*)carve((size_t)NT * 512 * 2);
  float* x1    = (float*)carve((size_t)NT * 512 * 4);
  float* xff   = f32a;

  auto wconv = [&](const float* W, u16* Wt, int K, int N) {
    convert_w_t<<<(K * N + 255) / 256, 256, 0, stream>>>(W, Wt, K, N);
  };
  wconv(Wff1, wt_ff1, 512, 2048);
  wconv(Wff2, wt_ff2, 2048, 512);
  wconv(Wq,   wt_qkv + (size_t)0 * 512 * 512, 512, 512);
  wconv(Wk,   wt_qkv + (size_t)1 * 512 * 512, 512, 512);
  wconv(Wv,   wt_qkv + (size_t)2 * 512 * 512, 512, 512);
  wconv(Wo,   wt_o,   512, 512);
  wconv(Wcp,  wt_cp,  512, 1024);
  wconv(Wco,  wt_co,  1024, 512);
  wconv(Wf1,  wt_f1,  512, 2048);
  wconv(Wf2,  wt_f2,  2048, 512);
  concat3<<<6, 256, 0, stream>>>(bq, bk, bv, bqkv);

  // --- Macaron FFN (pre) -> xff ---
  ln_bf16<<<NT / 8, 256, 0, stream>>>(x, g1, b1, lnbuf);
  gemm_wmma<EPI_RELU_BF16><<<dim3(2048 / 128, NT / 128), 256, 0, stream>>>(
      lnbuf, 512, wt_ff1, bff1, hidb, 2048, nullptr, nullptr, 512);
  gemm_wmma<EPI_F32><<<dim3(512 / 128, NT / 128), 256, 0, stream>>>(
      hidb, 2048, wt_ff2, bff2, xff, 512, nullptr, nullptr, 2048);

  // --- causal MHA (fused QKV projection) ---
  ln_bf16<<<NT / 8, 256, 0, stream>>>(x, ga, ba, lnbuf);
  gemm_wmma<EPI_BIAS_BF16><<<dim3(1536 / 128, NT / 128), 256, 0, stream>>>(
      lnbuf, 512, wt_qkv, bqkv, qkvb, 1536, nullptr, nullptr, 512);
  transpose_v_tiled<<<1024, 256, 0, stream>>>(qkvb, vtb);
  flash_attn<<<dim3(8, 64), 256, 0, stream>>>(qkvb, vtb, ctxb);
  // out-proj fused with residual combine: x1 = x + 0.5*xff + (ctx*Wo + bo)
  gemm_wmma<EPI_COMBINE_F32><<<dim3(512 / 128, NT / 128), 256, 0, stream>>>(
      ctxb, 512, wt_o, bo, x1, 512, x, xff, 512);

  // --- conv module ---
  ln_bf16<<<NT / 8, 256, 0, stream>>>(x1, gc, bc, lnbuf);
  gemm_wmma<EPI_GELU_F32><<<dim3(1024 / 128, NT / 128), 256, 0, stream>>>(
      lnbuf, 512, wt_cp, bcp, f32a, 1024, nullptr, nullptr, 512);
  dwconv_bn_swish<<<(NT * 1024 / 4) / 256, 256, 0, stream>>>(
      f32a, dwk, dwb, bng, bnb, bnm, bnv, hidb);
  gemm_wmma<EPI_RES_F32><<<dim3(512 / 128, NT / 128), 256, 0, stream>>>(
      hidb, 1024, wt_co, bco, x1, 512, x1, nullptr, 1024);

  // --- FFN (post) -> d_out ---
  ln_bf16<<<NT / 8, 256, 0, stream>>>(x1, g2, b2, lnbuf);
  gemm_wmma<EPI_RELU_BF16><<<dim3(2048 / 128, NT / 128), 256, 0, stream>>>(
      lnbuf, 512, wt_f1, bf1, hidb, 2048, nullptr, nullptr, 512);
  gemm_wmma<EPI_RES_F32><<<dim3(512 / 128, NT / 128), 256, 0, stream>>>(
      hidb, 2048, wt_f2, bf2, (float*)d_out, 512, x1, nullptr, 2048);
}
